// MixtureMLP_14139032338700
// MI455X (gfx1250) — matrix operand order repaired and checked by reading, not compiled, not run on hardware
//
#include <hip/hip_runtime.h>
#include <hip/hip_bf16.h>
#include <math.h>

// Problem constants (match reference)
#define NEXP 8
#define DMOD 1024
#define HDIM 4096
#define NTOK 16384

typedef __attribute__((ext_vector_type(16))) __bf16 v16bf;
typedef __attribute__((ext_vector_type(8)))  __bf16 v8bf;
typedef __attribute__((ext_vector_type(4)))  __bf16 v4bf;
typedef __attribute__((ext_vector_type(8)))  float  v8f;

// ---- CDNA5 async load-to-LDS path (guarded; falls back to manual staging) ----
#if __has_builtin(__builtin_amdgcn_global_load_async_to_lds_b128)
#define HAS_ASYNC_LDS 1
typedef int v4i_gnu __attribute__((vector_size(16)));
typedef __attribute__((address_space(1))) v4i_gnu* gas1_t;
typedef __attribute__((address_space(3))) v4i_gnu* las3_t;
#else
#define HAS_ASYNC_LDS 0
#endif

__device__ __forceinline__ void copy16_to_lds(const __bf16* g, __bf16* l) {
#if HAS_ASYNC_LDS
    __builtin_amdgcn_global_load_async_to_lds_b128((gas1_t)g, (las3_t)l, 0, 0);
#else
    *(v8bf*)l = *(const v8bf*)g;
#endif
}

__device__ __forceinline__ void wait_async_lds() {
#if HAS_ASYNC_LDS
#if __has_builtin(__builtin_amdgcn_s_wait_asynccnt)
    __builtin_amdgcn_s_wait_asynccnt(0);
#else
    asm volatile("s_wait_asynccnt 0x0" ::: "memory");
#endif
#endif
}

__device__ __forceinline__ v8f v8f_zero() {
    v8f z;
#pragma unroll
    for (int i = 0; i < 8; i++) z[i] = 0.0f;
    return z;
}

// A fragment (16x32 bf16, ISA 7.12.2): lane M = lane&15, half h = lane>>4.
// elements j=0..7 -> K = 8h+j ; j=8..15 -> K = 16+8h+(j-8). Row-major [M][K] LDS.
__device__ __forceinline__ v16bf ldfragA(const __bf16* row, int h) {
    v8bf lo = *(const v8bf*)(row + 8 * h);
    v8bf hi = *(const v8bf*)(row + 16 + 8 * h);
    v16bf f;
#pragma unroll
    for (int i = 0; i < 8; i++) { f[i] = lo[i]; f[i + 8] = hi[i]; }
    return f;
}

// B fragment (32x16 bf16): lane N = lane&15, half h = lane>>4.
// elements j=0..7 -> K = 8h+j ; j=8..15 -> K = 16+8h+(j-8). LDS holds B transposed: [N][K].
__device__ __forceinline__ v16bf ldfragB(const __bf16* row, int h) {
    v8bf lo = *(const v8bf*)(row + 8 * h);
    v8bf hi = *(const v8bf*)(row + 16 + 8 * h);
    v16bf f;
#pragma unroll
    for (int i = 0; i < 8; i++) { f[i] = lo[i]; f[i + 8] = hi[i]; }
    return f;
}

__device__ __forceinline__ v8f wmma_bf16(v16bf a, v16bf b, v8f c) {
    return __builtin_amdgcn_wmma_f32_16x16x32_bf16(false, a, false, b, (short)0, c,
                                                   false, false);
}

// ---------------------------------------------------------------- routing ---
__global__ void moe_zero_counts(int* counts) {
    if (threadIdx.x < NEXP) counts[threadIdx.x] = 0;
}

__global__ void moe_hist(const int* __restrict__ at, int* __restrict__ counts) {
    int i = blockIdx.x * blockDim.x + threadIdx.x;
    if (i < NTOK) atomicAdd(&counts[at[i]], 1);
}

__global__ void moe_scan(const int* __restrict__ counts, int* __restrict__ offsets,
                         int* __restrict__ cursors) {
    int acc = 0;
    for (int e = 0; e < NEXP; e++) {
        offsets[e] = acc;
        cursors[e] = acc;
        acc += counts[e];
    }
    offsets[NEXP] = acc;
}

__global__ void moe_scatter(const int* __restrict__ at, int* __restrict__ cursors,
                            int* __restrict__ perm) {
    int i = blockIdx.x * blockDim.x + threadIdx.x;
    if (i < NTOK) {
        int pos = atomicAdd(&cursors[at[i]], 1);
        perm[pos] = i;
    }
}

// gather x rows in sorted order, fp32 -> bf16
__global__ void moe_gather_x(const float* __restrict__ x, const int* __restrict__ perm,
                             __bf16* __restrict__ xb) {
    int gid = blockIdx.x * blockDim.x + threadIdx.x;   // NTOK * DMOD/4 threads
    int s = gid >> 8;                                   // DMOD/4 == 256
    int j = gid & 255;
    int tok = perm[s];
    float4 v = ((const float4*)x)[(size_t)tok * 256 + j];
    v4bf o;
    o[0] = (__bf16)v.x; o[1] = (__bf16)v.y; o[2] = (__bf16)v.z; o[3] = (__bf16)v.w;
    *(v4bf*)(xb + (size_t)s * DMOD + 4 * j) = o;
}

// tiled transpose + fp32->bf16:  in[e][R][C] -> out[e][C][R]
__global__ void moe_transpose_bf16(const float* __restrict__ in, __bf16* __restrict__ out,
                                   int R, int C) {
    __shared__ float tile[32][33];
    size_t base = (size_t)blockIdx.z * (size_t)R * (size_t)C;
    int c0 = blockIdx.x * 32, r0 = blockIdx.y * 32;
    int tx = threadIdx.x, ty = threadIdx.y;   // 32 x 8
#pragma unroll
    for (int i = 0; i < 4; i++)
        tile[ty + 8 * i][tx] = in[base + (size_t)(r0 + ty + 8 * i) * C + c0 + tx];
    __syncthreads();
#pragma unroll
    for (int i = 0; i < 4; i++)
        out[base + (size_t)(c0 + ty + 8 * i) * R + r0 + tx] = (__bf16)tile[tx][ty + 8 * i];
}

// ------------------------------------------------------------------ GEMM 1 ---
// act[s, col] = (x@Wp_proj + bp_p) * silu(x@Wp_gate + bp_g)
// block 64x128, K-step 32, double-buffered LDS, async loads to LDS.
__global__ __launch_bounds__(256) void moe_gemm1(
    const __bf16* __restrict__ xb,     // [NTOK][DMOD] sorted
    const __bf16* __restrict__ wpT,    // [E][2H][DMOD]
    const float*  __restrict__ bp,     // [E][2H]
    __bf16*       __restrict__ act,    // [NTOK][HDIM] sorted
    const int*    __restrict__ offsets) {
    int e = blockIdx.z;
    int seg0 = offsets[e];
    int segN = offsets[e + 1] - seg0;
    int row0 = blockIdx.x * 64;
    if (row0 >= segN) return;
    int col0 = blockIdx.y * 128;   // within [0, HDIM)

    __shared__ __align__(16) __bf16 As[2][64][40];
    __shared__ __align__(16) __bf16 Bp[2][128][40];
    __shared__ __align__(16) __bf16 Bg[2][128][40];

    int tid = threadIdx.x;
    int lane = tid & 31, wave = tid >> 5;
    int wr = wave >> 1, wc = wave & 1;     // 4 row-tiles x 2 col-halves
    int hh = lane >> 4, ln = lane & 15;

    v8f accP[4], accG[4];
#pragma unroll
    for (int i = 0; i < 4; i++) { accP[i] = v8f_zero(); accG[i] = v8f_zero(); }

    const __bf16* wpe   = wpT + (size_t)e * (size_t)(2 * HDIM) * DMOD;
    const __bf16* wproj = wpe + (size_t)col0 * DMOD;
    const __bf16* wgate = wpe + (size_t)(HDIM + col0) * DMOD;

    int ar  = tid >> 2, ach = (tid & 3) * 8;     // A: 64 rows x 4 chunks of 8
    int brn = tid >> 1, bch = (tid & 1) * 16;    // B: 128 rows x 2 chunks of 16
    int sA  = row0 + ar;
    size_t aRow = (size_t)(seg0 + (sA < segN ? sA : segN - 1)) * DMOD;

    auto issue = [&](int kt, int b) {
        int kb = kt * 32;
        const __bf16* ga = xb + aRow + kb + ach;
        const __bf16* sp = wproj + (size_t)brn * DMOD + kb + bch;
        const __bf16* sg = wgate + (size_t)brn * DMOD + kb + bch;
        copy16_to_lds(ga,     &As[b][ar][ach]);
        copy16_to_lds(sp,     &Bp[b][brn][bch]);
        copy16_to_lds(sp + 8, &Bp[b][brn][bch + 8]);
        copy16_to_lds(sg,     &Bg[b][brn][bch]);
        copy16_to_lds(sg + 8, &Bg[b][brn][bch + 8]);
        __builtin_prefetch(ga + 32, 0, 3);     // WGP-scope prefetch of tile kt+1
        __builtin_prefetch(sp + 32, 0, 3);
        __builtin_prefetch(sg + 32, 0, 3);
    };

    issue(0, 0);
    int buf = 0;
    for (int kt = 0; kt < DMOD / 32; kt++) {
        wait_async_lds();
        __syncthreads();
        if (kt + 1 < DMOD / 32) issue(kt + 1, buf ^ 1);

        v16bf a = ldfragA(&As[buf][16 * wr + ln][0], hh);
#pragma unroll
        for (int ct = 0; ct < 4; ct++) {
            int c = 64 * wc + 16 * ct + ln;
            v16bf fbP = ldfragB(&Bp[buf][c][0], hh);
            v16bf fbG = ldfragB(&Bg[buf][c][0], hh);
            accP[ct] = wmma_bf16(a, fbP, accP[ct]);
            accG[ct] = wmma_bf16(a, fbG, accG[ct]);
        }
        buf ^= 1;
    }

#pragma unroll
    for (int ct = 0; ct < 4; ct++) {
        int c = col0 + 64 * wc + 16 * ct + ln;
        float bpp = bp[(size_t)e * 2 * HDIM + c];
        float bpg = bp[(size_t)e * 2 * HDIM + HDIM + c];
#pragma unroll
        for (int i = 0; i < 8; i++) {
            int m = 16 * wr + i + 8 * hh;
            int s = row0 + m;
            if (s < segN) {
                float p = accP[ct][i] + bpp;
                float g = accG[ct][i] + bpg;
                float sg = g / (1.0f + __expf(-g));   // silu(g)
                act[(size_t)(seg0 + s) * HDIM + c] = (__bf16)(p * sg);
            }
        }
    }
}

// ------------------------------------------------------------------ GEMM 2 ---
// out[tok, d] = x[tok, d] + act@Wo + bo ; scatter by perm.
__global__ __launch_bounds__(256) void moe_gemm2(
    const __bf16* __restrict__ act,    // [NTOK][HDIM] sorted
    const __bf16* __restrict__ woT,    // [E][DMOD][HDIM]
    const float*  __restrict__ bo,     // [E][DMOD]
    const float*  __restrict__ x,      // [NTOK][DMOD] original order
    const int*    __restrict__ perm,
    const int*    __restrict__ offsets,
    float*        __restrict__ out) {
    int e = blockIdx.z;
    int seg0 = offsets[e];
    int segN = offsets[e + 1] - seg0;
    int row0 = blockIdx.x * 64;
    if (row0 >= segN) return;
    int col0 = blockIdx.y * 128;   // within [0, DMOD)

    __shared__ __align__(16) __bf16 As[2][64][40];
    __shared__ __align__(16) __bf16 Bs[2][128][40];

    int tid = threadIdx.x;
    int lane = tid & 31, wave = tid >> 5;
    int wr = wave >> 1, wc = wave & 1;
    int hh = lane >> 4, ln = lane & 15;

    v8f acc[4];
#pragma unroll
    for (int i = 0; i < 4; i++) acc[i] = v8f_zero();

    const __bf16* woe = woT + (size_t)e * (size_t)DMOD * HDIM + (size_t)col0 * HDIM;

    int ar  = tid >> 2, ach = (tid & 3) * 8;
    int brn = tid >> 1, bch = (tid & 1) * 16;
    int sA  = row0 + ar;
    size_t aRow = (size_t)(seg0 + (sA < segN ? sA : segN - 1)) * HDIM;

    auto issue = [&](int kt, int b) {
        int kb = kt * 32;
        const __bf16* ga = act + aRow + kb + ach;
        const __bf16* sb = woe + (size_t)brn * HDIM + kb + bch;
        copy16_to_lds(ga,     &As[b][ar][ach]);
        copy16_to_lds(sb,     &Bs[b][brn][bch]);
        copy16_to_lds(sb + 8, &Bs[b][brn][bch + 8]);
        __builtin_prefetch(ga + 32, 0, 3);
        __builtin_prefetch(sb + 32, 0, 3);
    };

    issue(0, 0);
    int buf = 0;
    for (int kt = 0; kt < HDIM / 32; kt++) {
        wait_async_lds();
        __syncthreads();
        if (kt + 1 < HDIM / 32) issue(kt + 1, buf ^ 1);

        v16bf a = ldfragA(&As[buf][16 * wr + ln][0], hh);
#pragma unroll
        for (int ct = 0; ct < 4; ct++) {
            v16bf b = ldfragB(&Bs[buf][64 * wc + 16 * ct + ln][0], hh);
            acc[ct] = wmma_bf16(a, b, acc[ct]);
        }
        buf ^= 1;
    }

#pragma unroll
    for (int i = 0; i < 8; i++) {
        int m = 16 * wr + i + 8 * hh;
        int s = row0 + m;
        if (s < segN) {
            int tok = perm[seg0 + s];
#pragma unroll
            for (int ct = 0; ct < 4; ct++) {
                int c = col0 + 64 * wc + 16 * ct + ln;
                out[(size_t)tok * DMOD + c] =
                    x[(size_t)tok * DMOD + c] + acc[ct][i] + bo[(size_t)e * DMOD + c];
            }
        }
    }
}

// ---------------------------------------------------------------- RMSNorm ---
__global__ void moe_rmsnorm(float* __restrict__ out, const float* __restrict__ nw) {
    int row = blockIdx.x, tid = threadIdx.x;         // 256 threads, 4 floats each
    float4* p = (float4*)(out + (size_t)row * DMOD);
    float4 v = p[tid];
    __shared__ float red[256];
    red[tid] = v.x * v.x + v.y * v.y + v.z * v.z + v.w * v.w;
    __syncthreads();
    for (int off = 128; off > 0; off >>= 1) {
        if (tid < off) red[tid] += red[tid + off];
        __syncthreads();
    }
    float scale = rsqrtf(red[0] * (1.0f / DMOD) + 1e-6f);
    float4 w = ((const float4*)nw)[tid];
    v.x *= scale * w.x; v.y *= scale * w.y; v.z *= scale * w.z; v.w *= scale * w.w;
    p[tid] = v;
}

// ----------------------------------------------------------------- launch ---
extern "C" void kernel_launch(void* const* d_in, const int* in_sizes, int n_in,
                              void* d_out, int out_size, void* d_ws, size_t ws_size,
                              hipStream_t stream) {
    const float* x   = (const float*)d_in[0];
    const int*   at  = (const int*)d_in[1];
    const float* Wp  = (const float*)d_in[2];
    const float* bp  = (const float*)d_in[3];
    const float* Wo  = (const float*)d_in[4];
    const float* bo  = (const float*)d_in[5];
    const float* nw  = (const float*)d_in[6];
    float* out = (float*)d_out;

    char* w = (char*)d_ws;
    int* counts  = (int*)w;            // 8
    int* cursors = counts + 8;         // 8
    int* offsets = counts + 16;        // 9
    int* perm    = counts + 32;        // NTOK
    __bf16* xb  = (__bf16*)(w + 65792);
    __bf16* wpT = (__bf16*)(w + 65792 + 33554432ull);
    __bf16* woT = (__bf16*)(w + 65792 + 33554432ull + 134217728ull);
    __bf16* actb = (__bf16*)(w + 65792 + 33554432ull + 134217728ull + 67108864ull);
    // total workspace use: ~352 MiB

    moe_zero_counts<<<1, 32, 0, stream>>>(counts);
    moe_hist<<<NTOK / 256, 256, 0, stream>>>(at, counts);
    moe_scan<<<1, 1, 0, stream>>>(counts, offsets, cursors);
    moe_scatter<<<NTOK / 256, 256, 0, stream>>>(at, cursors, perm);
    moe_gather_x<<<(NTOK * (DMOD / 4)) / 256, 256, 0, stream>>>(x, perm, xb);
    moe_transpose_bf16<<<dim3(2 * HDIM / 32, DMOD / 32, NEXP), dim3(32, 8), 0, stream>>>(
        Wp, wpT, DMOD, 2 * HDIM);
    moe_transpose_bf16<<<dim3(DMOD / 32, HDIM / 32, NEXP), dim3(32, 8), 0, stream>>>(
        Wo, woT, HDIM, DMOD);
    moe_gemm1<<<dim3(NTOK / 64, HDIM / 128, NEXP), 256, 0, stream>>>(xb, wpT, bp, actb,
                                                                     offsets);
    moe_gemm2<<<dim3(NTOK / 64, DMOD / 128, NEXP), 256, 0, stream>>>(actb, woT, bo, x,
                                                                     perm, offsets, out);
    moe_rmsnorm<<<NTOK, 256, 0, stream>>>(out, nw);
}
